// PointPillarScatter_Mix_Memory_54211077210393
// MI455X (gfx1250) — matrix operand order repaired and checked by reading, not compiled, not run on hardware
//
#include <hip/hip_runtime.h>
#include <hip/hip_bf16.h>

#define BATCH 2
#define PNUM  6000
#define NPT   4096
#define DIM   64
#define KTOP  4
#define MSLOT 1024
#define NXg   432
#define NYg   496
#define GRID  (NXg * NYg)          // 214272
#define NBEV  128
#define SHRINK_TH 0.0025f

typedef __attribute__((ext_vector_type(16))) _Float16 v16h;
typedef __attribute__((ext_vector_type(8)))  float    v8f;

__device__ __forceinline__ float red_max16(float v) {
  #pragma unroll
  for (int m = 1; m < 16; m <<= 1) v = fmaxf(v, __shfl_xor(v, m, 32));
  return v;
}
__device__ __forceinline__ float red_sum16(float v) {
  #pragma unroll
  for (int m = 1; m < 16; m <<= 1) v += __shfl_xor(v, m, 32);
  return v;
}

// Branchless insert of (v, gi) into a descending-sorted top-4 list.
// Conditional-swap chain -> v_cmp + v_cndmask only, no exec-mask divergence.
__device__ __forceinline__ void insert4(float v, int gi,
    float& t0, int& i0, float& t1, int& i1,
    float& t2, int& i2, float& t3, int& i3) {
  bool c; float fv; int fi;
  c = v > t0; fv = c ? t0 : v; fi = c ? i0 : gi;
  t0 = c ? v : t0; i0 = c ? gi : i0; v = fv; gi = fi;
  c = v > t1; fv = c ? t1 : v; fi = c ? i1 : gi;
  t1 = c ? v : t1; i1 = c ? gi : i1; v = fv; gi = fi;
  c = v > t2; fv = c ? t2 : v; fi = c ? i2 : gi;
  t2 = c ? v : t2; i2 = c ? gi : i2; v = fv; gi = fi;
  c = v > t3;
  t3 = c ? v : t3; i3 = c ? gi : i3;
}

// ---------------------------------------------------------------------------
// Kernel 1: logits = pillars @ points^T via WMMA, fused per-row top-4.
// softmax is monotonic per row -> top-k of logits == top-k of softmax scores.
// One WG = 16 pillars; 4 waves split the 4096 points (1024 each, 64 tiles).
// All 32 lanes scan: lane l16 owns row l16, lane-half owns an 8-column slice;
// halves merge via shfl_xor(16) at the end (symmetric -> identical lists).
// ---------------------------------------------------------------------------
__global__ __launch_bounds__(128) void topk_kernel(
    const float* __restrict__ pillars, const float* __restrict__ points,
    int* __restrict__ idx_out) {
  __shared__ float stage[4][16][17];
  __shared__ float mval[4][16][KTOP];
  __shared__ int   midx[4][16][KTOP];

  const int wg   = blockIdx.x;
  const int b    = wg / (PNUM / 16);
  const int ptile= wg % (PNUM / 16);
  const int wave = threadIdx.x >> 5;
  const int lane = threadIdx.x & 31;
  const int half = lane >> 4;
  const int l16  = lane & 15;

  const float* prow  = pillars + ((size_t)b * PNUM + (size_t)ptile * 16) * DIM;
  const float* pbase = points  + (size_t)b * NPT * DIM;

  // A fragments (16x64 pillar tile -> two K=32 frags). A layout used here (contiguous k, same as B):
  // lanes 0-15: row=l16, K = {k0..k0+15} with k0 = half*16.
  v16h a0, a1;
  {
    const float* s = prow + (size_t)l16 * DIM;
    const int k0 = half * 16;
    #pragma unroll
    for (int i = 0; i < 8; ++i) {
      a0[i]     = (_Float16)s[k0 + i];
      a0[8 + i] = (_Float16)s[k0 + 8 + i];
      a1[i]     = (_Float16)s[32 + k0 + i];
      a1[8 + i] = (_Float16)s[32 + k0 + 8 + i];
    }
  }

  float tv0 = -1e30f, tv1 = -1e30f, tv2 = -1e30f, tv3 = -1e30f;
  int   ti0 = 0, ti1 = 0, ti2 = 0, ti3 = 0;

  const int nbase_w = wave * (NPT / 4);
  for (int nt = 0; nt < (NPT / 4) / 16; ++nt) {
    const int nb = nbase_w + nt * 16;
    // B fragments: lane column = nb + l16; K half selected by lane half.
    v16h b0, b1;
    {
      const float* p = pbase + (size_t)(nb + l16) * DIM + half * 16;
      __builtin_prefetch(p + 16 * DIM, 0, 1);   // global_prefetch_b8: next tile
      #pragma unroll
      for (int i = 0; i < 16; ++i) { b0[i] = (_Float16)p[i]; b1[i] = (_Float16)p[32 + i]; }
    }
    v8f c = {};
    c = __builtin_amdgcn_wmma_f32_16x16x32_f16(false, a0, false, b0, (short)0, c, false, false);
    c = __builtin_amdgcn_wmma_f32_16x16x32_f16(false, a1, false, b1, (short)0, c, false, false);

    // C layout: lane holds column l16, rows r+8*half across 8 VGPRs.
    #pragma unroll
    for (int r = 0; r < 8; ++r) stage[wave][r + 8 * half][l16] = c[r];
    __builtin_amdgcn_wave_barrier();
    asm volatile("s_wait_dscnt 0" ::: "memory");

    // every lane scans its row / 8-column slice; fully branchless
    #pragma unroll
    for (int j = 0; j < 8; ++j) {
      const int col = half * 8 + j;
      insert4(stage[wave][l16][col], nb + col,
              tv0, ti0, tv1, ti1, tv2, ti2, tv3, ti3);
    }
    __builtin_amdgcn_wave_barrier();
  }

  // merge the two column-halves of each row via shfl_xor(16); symmetric merge
  {
    float pv0 = __shfl_xor(tv0, 16, 32), pv1 = __shfl_xor(tv1, 16, 32);
    float pv2 = __shfl_xor(tv2, 16, 32), pv3 = __shfl_xor(tv3, 16, 32);
    int   pi0 = __shfl_xor(ti0, 16, 32), pi1 = __shfl_xor(ti1, 16, 32);
    int   pi2 = __shfl_xor(ti2, 16, 32), pi3 = __shfl_xor(ti3, 16, 32);
    insert4(pv0, pi0, tv0, ti0, tv1, ti1, tv2, ti2, tv3, ti3);
    insert4(pv1, pi1, tv0, ti0, tv1, ti1, tv2, ti2, tv3, ti3);
    insert4(pv2, pi2, tv0, ti0, tv1, ti1, tv2, ti2, tv3, ti3);
    insert4(pv3, pi3, tv0, ti0, tv1, ti1, tv2, ti2, tv3, ti3);
  }
  if (lane < 16) {
    mval[wave][lane][0]=tv0; mval[wave][lane][1]=tv1; mval[wave][lane][2]=tv2; mval[wave][lane][3]=tv3;
    midx[wave][lane][0]=ti0; midx[wave][lane][1]=ti1; midx[wave][lane][2]=ti2; midx[wave][lane][3]=ti3;
  }
  __syncthreads();
  if (wave == 0 && lane < 16) {
    float v0=-1e30f,v1=-1e30f,v2=-1e30f,v3=-1e30f; int i0=0,i1=0,i2=0,i3=0;
    #pragma unroll
    for (int w = 0; w < 4; ++w)
      #pragma unroll
      for (int j = 0; j < KTOP; ++j)
        insert4(mval[w][lane][j], midx[w][lane][j],
                v0, i0, v1, i1, v2, i2, v3, i3);
    int* o = idx_out + ((size_t)b * PNUM + (size_t)ptile * 16 + lane) * KTOP;
    o[0]=i0; o[1]=i1; o[2]=i2; o[3]=i3;
  }
}

// ---------------------------------------------------------------------------
// Kernel 2: fused MemoryUnit. mem_w lives in LDS as f16 (128 KB; dynamic LDS,
// CDNA5 WGP has 320 KB). Per wave: 16 rows of x = points[topk].
// Pass 1: logits GEMM (WMMA) + online softmax stats (shfl reductions).
// Pass 2: recompute logits, att=exp/sum, hard-shrink == threshold, stage keep
// tiles to LDS as f16, reload as A-fragments, WMMA att@mem_w. L1 renorm is a
// per-row scalar divide at the end (commutes with the GEMM).
// ---------------------------------------------------------------------------
__global__ __launch_bounds__(128) void memory_kernel(
    const float* __restrict__ points, const int* __restrict__ idx_in,
    const float* __restrict__ mem_w, float* __restrict__ mem_out) {
  extern __shared__ char smem[];
  _Float16* memw  = (_Float16*)smem;                              // [1024][64]
  _Float16* stg   = (_Float16*)(smem + MSLOT * DIM * 2);          // [4][16][32]

  for (int i = threadIdx.x; i < MSLOT * DIM; i += 128)
    memw[i] = (_Float16)mem_w[i];
  __syncthreads();

  const int wg   = blockIdx.x;
  const int rows_per_batch = PNUM * KTOP;                          // 24000
  const int b    = wg / (rows_per_batch / 64);
  const int rb   = (wg % (rows_per_batch / 64)) * 64;
  const int wave = threadIdx.x >> 5;
  const int lane = threadIdx.x & 31;
  const int half = lane >> 4;
  const int l16  = lane & 15;
  const int rwb  = rb + wave * 16;

  // gather this lane's x row: row -> (pillar, k) -> point index
  const int myrow = rwb + l16;
  const int pil = myrow >> 2, kk = myrow & 3;
  const int pt  = idx_in[((size_t)b * PNUM + pil) * KTOP + kk];
  const float* src = points + ((size_t)b * NPT + pt) * DIM;

  v16h a0, a1;
  {
    const int k0 = half * 16;
    #pragma unroll
    for (int i = 0; i < 8; ++i) {
      a0[i]     = (_Float16)src[k0 + i];
      a0[8 + i] = (_Float16)src[k0 + 8 + i];
      a1[i]     = (_Float16)src[32 + k0 + i];
      a1[8 + i] = (_Float16)src[32 + k0 + 8 + i];
    }
  }

  float rmax[8], rsum[8];
  #pragma unroll
  for (int r = 0; r < 8; ++r) { rmax[r] = -1e30f; rsum[r] = 0.f; }

  // ---- pass 1: streaming softmax statistics over 64 column tiles ----
  for (int nt = 0; nt < MSLOT / 16; ++nt) {
    v16h b0, b1;
    const _Float16* wp = memw + (size_t)(nt * 16 + l16) * DIM + half * 16;
    #pragma unroll
    for (int i = 0; i < 16; ++i) { b0[i] = wp[i]; b1[i] = wp[32 + i]; }
    v8f c = {};
    c = __builtin_amdgcn_wmma_f32_16x16x32_f16(false, a0, false, b0, (short)0, c, false, false);
    c = __builtin_amdgcn_wmma_f32_16x16x32_f16(false, a1, false, b1, (short)0, c, false, false);
    #pragma unroll
    for (int r = 0; r < 8; ++r) {
      const float m  = red_max16(c[r]);
      const float nm = fmaxf(rmax[r], m);
      const float e  = red_sum16(__expf(c[r] - nm));
      rsum[r] = rsum[r] * __expf(rmax[r] - nm) + e;
      rmax[r] = nm;
    }
  }
  float inv[8];
  #pragma unroll
  for (int r = 0; r < 8; ++r) inv[r] = 1.0f / rsum[r];

  // ---- pass 2: att values + shrink, second GEMM (att @ mem_w) ----
  float l1p[8];
  #pragma unroll
  for (int r = 0; r < 8; ++r) l1p[r] = 0.f;
  v8f acc0 = {}, acc1 = {}, acc2 = {}, acc3 = {};
  _Float16* st = stg + (size_t)wave * 16 * 32;

  for (int nt = 0; nt < MSLOT / 16; nt += 2) {
    #pragma unroll
    for (int u = 0; u < 2; ++u) {
      const int t = nt + u;
      v16h b0, b1;
      const _Float16* wp = memw + (size_t)(t * 16 + l16) * DIM + half * 16;
      #pragma unroll
      for (int i = 0; i < 16; ++i) { b0[i] = wp[i]; b1[i] = wp[32 + i]; }
      v8f c = {};
      c = __builtin_amdgcn_wmma_f32_16x16x32_f16(false, a0, false, b0, (short)0, c, false, false);
      c = __builtin_amdgcn_wmma_f32_16x16x32_f16(false, a1, false, b1, (short)0, c, false, false);
      #pragma unroll
      for (int r = 0; r < 8; ++r) {
        const float att = __expf(c[r] - rmax[r]) * inv[r];
        // relu(a-l)*a/(|a-l|+eps) == (a > l ? a : 0)
        const float kp = (att > SHRINK_TH) ? att : 0.f;
        l1p[r] += kp;
        st[(size_t)(r + 8 * half) * 32 + u * 16 + l16] = (_Float16)kp;
      }
    }
    __builtin_amdgcn_wave_barrier();
    asm volatile("s_wait_dscnt 0" ::: "memory");

    // A fragment from staged keep tile (16 rows x K=32)
    v16h ka;
    {
      const int k0 = half * 16;
      #pragma unroll
      for (int i = 0; i < 8; ++i) {
        ka[i]     = st[(size_t)l16 * 32 + k0 + i];
        ka[8 + i] = st[(size_t)l16 * 32 + 8 + k0 + i];
      }
    }
    const int kg = nt * 16;     // global K offset into mem_w rows
    const int kq = half * 16;
    v16h kb0, kb1, kb2, kb3;
    #pragma unroll
    for (int i = 0; i < 16; ++i) {
      const _Float16* wr = memw + (size_t)(kg + kq + i) * DIM + l16;
      kb0[i] = wr[0]; kb1[i] = wr[16]; kb2[i] = wr[32]; kb3[i] = wr[48];
    }
    acc0 = __builtin_amdgcn_wmma_f32_16x16x32_f16(false, ka, false, kb0, (short)0, acc0, false, false);
    acc1 = __builtin_amdgcn_wmma_f32_16x16x32_f16(false, ka, false, kb1, (short)0, acc1, false, false);
    acc2 = __builtin_amdgcn_wmma_f32_16x16x32_f16(false, ka, false, kb2, (short)0, acc2, false, false);
    acc3 = __builtin_amdgcn_wmma_f32_16x16x32_f16(false, ka, false, kb3, (short)0, acc3, false, false);
    __builtin_amdgcn_wave_barrier();
  }

  // finalize: L1 renorm scalar, store out[16 rows x 64]
  float* orow = mem_out + ((size_t)b * rows_per_batch + rwb) * DIM;
  #pragma unroll
  for (int r = 0; r < 8; ++r) {
    const float l1 = red_sum16(l1p[r]);
    const float s  = 1.0f / (l1 + 1e-12f);
    const int row  = r + 8 * half;
    orow[(size_t)row * DIM +  0 + l16] = acc0[r] * s;
    orow[(size_t)row * DIM + 16 + l16] = acc1[r] * s;
    orow[(size_t)row * DIM + 32 + l16] = acc2[r] * s;
    orow[(size_t)row * DIM + 48 + l16] = acc3[r] * s;
  }
}

// ---------------------------------------------------------------------------
// Kernel 3: h = mem_out @ adapt_w^T  (adapt_w in LDS), + BN stats via atomics
// ---------------------------------------------------------------------------
__global__ __launch_bounds__(256) void adapt_kernel(
    const float* __restrict__ mem_out, const float* __restrict__ adapt_w,
    float* __restrict__ h_out, float* __restrict__ stats) {
  __shared__ float wsm[64 * 256];
  for (int i = threadIdx.x; i < 64 * 256; i += 256) wsm[i] = adapt_w[i];
  __syncthreads();

  const int wg = blockIdx.x;
  const int b  = wg / (PNUM / 4);
  const int p  = (wg % (PNUM / 4)) * 4 + (threadIdx.x >> 6);
  const int c  = threadIdx.x & 63;

  const float* mrow = mem_out + ((size_t)b * PNUM + p) * 256;
  const float* wrow = wsm + (size_t)c * 256;
  float acc = 0.f;
  #pragma unroll 8
  for (int j = 0; j < 256; ++j) acc += mrow[j] * wrow[j];
  h_out[((size_t)b * PNUM + p) * 64 + c] = acc;
  atomicAdd(&stats[b * 128 + c], acc);
  atomicAdd(&stats[b * 128 + 64 + c], acc * acc);
}

// ---------------------------------------------------------------------------
// Kernel 4: BN + ReLU + concat(pillars, h) scattered into dense BEV grid
// ---------------------------------------------------------------------------
__global__ void scatter_kernel(
    const float* __restrict__ pillars, const float* __restrict__ h,
    const float* __restrict__ stats, const float* __restrict__ gamma,
    const float* __restrict__ beta, const int* __restrict__ coords,
    float* __restrict__ out) {
  const int tid = blockIdx.x * blockDim.x + threadIdx.x;
  if (tid >= BATCH * PNUM * NBEV) return;
  const int c = tid & (NBEV - 1);
  const int p = (tid >> 7) % PNUM;
  const int b = tid / (PNUM * NBEV);
  const int* cr = coords + ((size_t)b * PNUM + p) * 3;     // (z, y, x)
  const int flat = cr[0] + cr[1] * NXg + cr[2];
  float v;
  if (c < 64) {
    v = pillars[((size_t)b * PNUM + p) * 64 + c];
  } else {
    const int cc = c - 64;
    const float mu  = stats[b * 128 + cc] * (1.0f / PNUM);
    const float var = stats[b * 128 + 64 + cc] * (1.0f / PNUM) - mu * mu;
    const float x   = h[((size_t)b * PNUM + p) * 64 + cc];
    v = (x - mu) * rsqrtf(var + 1e-3f) * gamma[cc] + beta[cc];
    v = v > 0.f ? v : 0.f;
  }
  out[(size_t)b * NBEV * GRID + (size_t)c * GRID + flat] = v;
}

__global__ void pind_kernel(const int* __restrict__ coords, float* __restrict__ out) {
  const int tid = blockIdx.x * blockDim.x + threadIdx.x;
  if (tid >= BATCH * PNUM) return;
  const int b = tid / PNUM, p = tid % PNUM;
  const int* cr = coords + ((size_t)b * PNUM + p) * 3;
  const int flat = cr[0] + cr[1] * NXg + cr[2];
  float* base = out + (size_t)BATCH * NBEV * GRID + (size_t)b * 3 * GRID;
  base[(size_t)0 * GRID + flat] = (float)cr[1];
  base[(size_t)1 * GRID + flat] = (float)cr[2];
  base[(size_t)2 * GRID + flat] = (float)cr[0];
}

extern "C" void kernel_launch(void* const* d_in, const int* in_sizes, int n_in,
                              void* d_out, int out_size, void* d_ws, size_t ws_size,
                              hipStream_t stream) {
  const float* pillars = (const float*)d_in[0];
  const float* points  = (const float*)d_in[1];
  const int*   coords  = (const int*)d_in[2];
  const float* adapt_w = (const float*)d_in[3];
  const float* mem_w   = (const float*)d_in[4];
  const float* gamma   = (const float*)d_in[5];
  const float* beta    = (const float*)d_in[6];
  float* out = (float*)d_out;

  char* ws = (char*)d_ws;
  int*   idx     = (int*)ws;                                  // 192 KB
  float* mem_out = (float*)(ws + (1u << 20));                 // 12.3 MB
  float* h       = (float*)(ws + 14u * (1u << 20));           // 3.1 MB
  float* stats   = (float*)(ws + 18u * (1u << 20));           // 1 KB

  hipMemsetAsync(stats, 0, (size_t)BATCH * 2 * 64 * sizeof(float), stream);
  hipMemsetAsync(out, 0, (size_t)out_size * sizeof(float), stream);

  topk_kernel<<<BATCH * (PNUM / 16), 128, 0, stream>>>(pillars, points, idx);

  const size_t smem = (size_t)MSLOT * DIM * 2 + 4 * 16 * 32 * 2;   // 132 KB
  memory_kernel<<<BATCH * ((PNUM * KTOP) / 64), 128, smem, stream>>>(
      points, idx, mem_w, mem_out);

  adapt_kernel<<<BATCH * (PNUM / 4), 256, 0, stream>>>(mem_out, adapt_w, h, stats);

  const int tot = BATCH * PNUM * NBEV;
  scatter_kernel<<<(tot + 255) / 256, 256, 0, stream>>>(pillars, h, stats, gamma, beta, coords, out);

  const int tot2 = BATCH * PNUM;
  pind_kernel<<<(tot2 + 255) / 256, 256, 0, stream>>>(coords, out);
}
